// RGCLayer_24309514895849
// MI455X (gfx1250) — compile-verified, matrix operand order checked by reading, not run on hardware
//
#include <hip/hip_runtime.h>

#define N_NODES 2048
#define DIM     512
#define HDIM    512
#define NPRED   51
#define NBASE   30

typedef __attribute__((ext_vector_type(16))) __bf16 bf16x16;
typedef __attribute__((ext_vector_type(8)))  float  f32x8;

union BfFrag { bf16x16 v; unsigned u[8]; };

// LDS row stride: 32 bf16 data padded to 40 bf16 (80B) -> 16B-aligned rows for
// async B128 writes, and 20-u32 stride -> conflict-free fragment gathers.
#define LSTR_BF  40
#define LSTR_U32 20

// VpT[p][h][d] = sum_b W_comp[p,b] * W[b*D+d, h]   (bf16, B-transposed layout)
__global__ void k_compV(const float* __restrict__ W, const float* __restrict__ Wc,
                        __bf16* __restrict__ VpT) {
  int idx = blockIdx.x * blockDim.x + threadIdx.x;
  int p   = idx / (HDIM * DIM);
  int rem = idx - p * (HDIM * DIM);
  int h   = rem / DIM;
  int d   = rem - h * DIM;
  float s = 0.f;
#pragma unroll
  for (int b = 0; b < NBASE; ++b)
    s += Wc[p * NBASE + b] * W[((size_t)b * DIM + d) * HDIM + h];
  VpT[idx] = (__bf16)s;
}

__global__ void k_init_out(const float* __restrict__ Bias, float* __restrict__ out) {
  int idx = blockIdx.x * blockDim.x + threadIdx.x;
  out[idx] = Bias[idx & (HDIM - 1)];
}

// 128(M) x 64(N) tile GEMM: C = Amat(fp32 [M][K]) x Bt(bf16 [Ncols][K])^T
// 4 waves; each wave owns 32 rows x 64 cols = 8 accumulators (2 A-frags x 4 B-frags).
// B tiles staged LDS-direct via global_load_async_to_lds_b128 (ASYNCcnt).
// ACCUM_F32=false: store bf16 TRANSPOSED  Cout[col][row] (ldc = M stride)
// ACCUM_F32=true : fp32 accumulate        Cout[row][col] += acc
template <bool ACCUM_F32>
__global__ __launch_bounds__(128)
void k_wmma_gemm(const float* __restrict__ Amat, int lda,
                 const __bf16* __restrict__ Bt, int ldb,
                 void* __restrict__ Cout, int ldc, int K) {
  __shared__ __align__(16) __bf16 ldsA[128 * LSTR_BF];
  __shared__ __align__(16) __bf16 ldsB[64 * LSTR_BF];

  const int tid  = threadIdx.x;
  const int lane = tid & 31;
  const int wv   = tid >> 5;          // wave 0..3 -> rows 32*wv..32*wv+31
  const int half = lane >> 4;         // 0: K {0..7,16..23}, 1: K {8..15,24..31}
  const int l16  = lane & 15;
  const int kb2  = half << 2;         // K pair-index base
  const int n0   = blockIdx.x * 64;   // output cols
  const int m0   = blockIdx.y * 128;  // output rows

  const unsigned ldsB_base = (unsigned)(uintptr_t)(void*)ldsB; // low 32b = LDS offset

  f32x8 acc[2][4] = {};

  for (int k0 = 0; k0 < K; k0 += 32) {
    // --- stage A tile 128x32: fp32 -> bf16 through VGPRs (needs convert) ---
#pragma unroll
    for (int i = 0; i < 32; ++i) {
      int idx = tid + i * 128;
      int r = idx >> 5, c = idx & 31;
      ldsA[r * LSTR_BF + c] = (__bf16)Amat[(size_t)(m0 + r) * lda + (k0 + c)];
    }
    // --- stage B tile 64x32 bf16: async global->LDS, 16B chunks, no VGPR round-trip ---
#pragma unroll
    for (int i = 0; i < 2; ++i) {
      int q = tid + i * 128;          // 256 chunks of 16B
      int r = q >> 2, c4 = q & 3;
      unsigned    dst = ldsB_base + (unsigned)(r * (LSTR_BF * 2) + c4 * 16);
      const void* src = (const char*)Bt + ((size_t)(n0 + r) * ldb + k0) * 2 + c4 * 16;
      asm volatile("global_load_async_to_lds_b128 %0, %1, off"
                   :: "v"(dst), "v"(src) : "memory");
    }
    asm volatile("s_wait_asynccnt 0x0" ::: "memory");
    __syncthreads();

    const unsigned* lA = (const unsigned*)ldsA;
    const unsigned* lB = (const unsigned*)ldsB;

    // two A fragments: rows 32*wv + l16 and +16
    BfFrag a0, a1;
    {
      int r0 = (32 * wv + l16) * LSTR_U32;
      int r1 = r0 + 16 * LSTR_U32;
#pragma unroll
      for (int j = 0; j < 8; ++j) {
        int k2 = (j < 4) ? (kb2 + j) : (8 + kb2 + (j - 4));
        a0.u[j] = lA[r0 + k2];
        a1.u[j] = lA[r1 + k2];
      }
    }
#pragma unroll
    for (int c = 0; c < 4; ++c) {
      BfFrag b;
      int rB = (16 * c + l16) * LSTR_U32;
#pragma unroll
      for (int j = 0; j < 8; ++j) {
        int k2 = (j < 4) ? (kb2 + j) : (8 + kb2 + (j - 4));
        b.u[j] = lB[rB + k2];
      }
      acc[0][c] = __builtin_amdgcn_wmma_f32_16x16x32_bf16(
          false, a0.v, false, b.v, (short)0, acc[0][c], false, false);
      acc[1][c] = __builtin_amdgcn_wmma_f32_16x16x32_bf16(
          false, a1.v, false, b.v, (short)0, acc[1][c], false, false);
    }
    __syncthreads();
  }

  // C/D layout: lane col = l16, VGPR j -> row j + 8*half within each 16x16 tile
  if constexpr (ACCUM_F32) {
    float* O = (float*)Cout;
#pragma unroll
    for (int i = 0; i < 2; ++i)
#pragma unroll
      for (int c = 0; c < 4; ++c) {
        int col = n0 + 16 * c + l16;
#pragma unroll
        for (int j = 0; j < 8; ++j) {
          int row = m0 + 32 * wv + 16 * i + j + 8 * half;
          O[(size_t)row * ldc + col] += acc[i][c][j];
        }
      }
  } else {
    __bf16* O = (__bf16*)Cout;
#pragma unroll
    for (int i = 0; i < 2; ++i)
#pragma unroll
      for (int c = 0; c < 4; ++c) {
        int col = n0 + 16 * c + l16;
#pragma unroll
        for (int j = 0; j < 8; ++j) {
          int row = m0 + 32 * wv + 16 * i + j + 8 * half;
          O[(size_t)col * ldc + row] = (__bf16)acc[i][c][j];  // transposed store
        }
      }
  }
}

extern "C" void kernel_launch(void* const* d_in, const int* in_sizes, int n_in,
                              void* d_out, int out_size, void* d_ws, size_t ws_size,
                              hipStream_t stream) {
  const float* vertex = (const float*)d_in[0];  // [N, D]
  const float* A      = (const float*)d_in[1];  // [P, N, N]
  const float* W      = (const float*)d_in[2];  // [NB*D, H]
  const float* Wc     = (const float*)d_in[3];  // [P, NB]
  const float* Bias   = (const float*)d_in[4];  // [H]
  float*       out    = (float*)d_out;          // [N, H]

  __bf16* VpT = (__bf16*)d_ws;                        // [P][H][D]  26.7 MB
  __bf16* ZpT = VpT + (size_t)NPRED * HDIM * DIM;     // [H][N]      2.1 MB

  k_compV<<<(NPRED * HDIM * DIM) / 256, 256, 0, stream>>>(W, Wc, VpT);
  k_init_out<<<(N_NODES * HDIM) / 256, 256, 0, stream>>>(Bias, out);

  for (int p = 0; p < NPRED; ++p) {
    // ZpT[h][m] = (vertex @ Vp)^T      M=N_NODES, Ncols=HDIM, K=DIM
    k_wmma_gemm<false><<<dim3(HDIM / 64, N_NODES / 128), 128, 0, stream>>>(
        vertex, DIM, VpT + (size_t)p * HDIM * DIM, DIM,
        (void*)ZpT, N_NODES, DIM);
    // out += A[p] @ Zp                 M=N_NODES, Ncols=HDIM, K=N_NODES
    k_wmma_gemm<true><<<dim3(HDIM / 64, N_NODES / 128), 128, 0, stream>>>(
        A + (size_t)p * N_NODES * N_NODES, N_NODES, ZpT, N_NODES,
        (void*)out, HDIM, N_NODES);
  }
}
